// SelfAttention1D_8546984919497
// MI455X (gfx1250) — matrix operand order repaired and checked by reading, not compiled, not run on hardware
//
#include <hip/hip_runtime.h>
#include <hip/hip_bf16.h>
#include <stdint.h>

#define NROWS 1024
#define DDIM  512

typedef __attribute__((ext_vector_type(16))) __bf16 v16bf;
typedef __attribute__((ext_vector_type(8)))  __bf16 v8bf;
typedef __attribute__((ext_vector_type(8)))  float  v8f;

// round-to-nearest-even f32 -> bf16, done with integer ops so we never rely
// on __bf16 scalar arithmetic support.
__device__ __forceinline__ unsigned short f32_to_bf16(float f) {
  unsigned int u = __float_as_uint(f);
  u += 0x7FFFu + ((u >> 16) & 1u);
  return (unsigned short)(u >> 16);
}

// ---------------------------------------------------------------------------
// Pass 1: convert x -> bf16 row-major, and Wq/Wk/Wv -> bf16 TRANSPOSED [N,K]
// so the WMMA B-fragment (lane = N column, elements = contiguous K) can be
// loaded with two b128 loads per lane.
// ---------------------------------------------------------------------------
__global__ __launch_bounds__(256) void convert_kernel(
    const float* __restrict__ x,
    const float* __restrict__ Wq,
    const float* __restrict__ Wk,
    const float* __restrict__ Wv,
    unsigned short* __restrict__ xb,    // [NROWS*DDIM]
    unsigned short* __restrict__ wts)   // 3 x [DDIM*DDIM], each transposed
{
  int idx = blockIdx.x * blockDim.x + threadIdx.x;
  if (idx < NROWS * DDIM) xb[idx] = f32_to_bf16(x[idx]);
  if (idx < DDIM * DDIM) {
    int k = idx / DDIM, n = idx % DDIM;
    int t = n * DDIM + k;               // transpose: [k][n] -> [n][k]
    wts[0 * DDIM * DDIM + t] = f32_to_bf16(Wq[idx]);
    wts[1 * DDIM * DDIM + t] = f32_to_bf16(Wk[idx]);
    wts[2 * DDIM * DDIM + t] = f32_to_bf16(Wv[idx]);
  }
}

// ---------------------------------------------------------------------------
// Pass 2: Q/K/V projections. One wave = one 16x16 output tile, K-loop in
// steps of 32 using v_wmma_f32_16x16x32_bf16. Fragment layouts follow the
// CDNA5 ISA 16-bit A (16x32) / B (32x16) / f32 C-D (16x16) tables:
//   A: lane<16 row=lane, elems 0..7 = K kb+0..7, elems 8..15 = K kb+16..23
//      lane>=16 row=lane-16, elems = K kb+8..15 / kb+24..31
//   B: lane gives N; elems 0..7 = K kb+8*half+0..7, elems 8..15 =
//      K kb+16+8*half+0..7 (same as A)  -> contiguous in transposed W
//   D: lane gives N; VGPR r -> M = (lane<16 ? r : 8+r)
// ---------------------------------------------------------------------------
__global__ __launch_bounds__(32) void proj_gemm_wmma(
    const unsigned short* __restrict__ xb,   // [NROWS, DDIM] bf16
    const unsigned short* __restrict__ wts,  // 3 x [DDIM, DDIM] bf16 (W^T)
    float* __restrict__ qkv)                 // 3 x [NROWS, DDIM] f32
{
  const int wi = blockIdx.z;  // 0=Q, 1=K, 2=V
  const unsigned short* wt = wts + (size_t)wi * DDIM * DDIM;
  float* out = qkv + (size_t)wi * NROWS * DDIM;

  const int lane = threadIdx.x;
  const int l16  = lane & 15;
  const int half = lane >> 4;   // 0: lanes 0-15, 1: lanes 16-31

  const unsigned short* aptr = xb + (size_t)(blockIdx.x * 16 + l16) * DDIM;
  const unsigned short* bptr = wt + (size_t)(blockIdx.y * 16 + l16) * DDIM;

  const int koff = half * 8;    // A-fragment K phase per lane-half
  const int boff = half * 8;    // B-fragment K phase per lane-half

  v8f acc = {};
  #pragma unroll 4
  for (int kb = 0; kb < DDIM; kb += 32) {
    v8bf a0 = *(const v8bf*)(aptr + kb + koff);
    v8bf a1 = *(const v8bf*)(aptr + kb + 16 + koff);
    v8bf b0 = *(const v8bf*)(bptr + kb + boff);
    v8bf b1 = *(const v8bf*)(bptr + kb + boff + 16);
    v16bf a, b;
    #pragma unroll
    for (int e = 0; e < 8; ++e) {
      a[e] = a0[e]; a[e + 8] = a1[e];
      b[e] = b0[e]; b[e + 8] = b1[e];
    }
    acc = __builtin_amdgcn_wmma_f32_16x16x32_bf16(
        /*neg_a=*/false, a, /*neg_b=*/false, b,
        /*c_mod=*/(short)0, acc, /*reuse_a=*/false, /*reuse_b=*/false);
  }

  float* optr = out + (size_t)(blockIdx.x * 16 + half * 8) * DDIM
                    + (size_t)(blockIdx.y * 16 + l16);
  #pragma unroll
  for (int r = 0; r < 8; ++r) optr[(size_t)r * DDIM] = acc[r];
}

// ---------------------------------------------------------------------------
// Pass 3: rank-1 softmax attention. One block (512 thr = 16 wave32) per batch
// row b. k,v rows staged in LDS (conflict-free broadcast reads).
//   out[b,i] = sum_j exp(q_i*k_j - m_i) * v_j / sum_j exp(q_i*k_j - m_i)
// with the exact row max m_i = max(q_i*kmax, q_i*kmin).
// ---------------------------------------------------------------------------
__global__ __launch_bounds__(DDIM) void attn_kernel(
    const float* __restrict__ qkv, float* __restrict__ out)
{
  const float* Q = qkv;
  const float* K = qkv + (size_t)NROWS * DDIM;
  const float* V = qkv + (size_t)2 * NROWS * DDIM;

  __shared__ float sk[DDIM];
  __shared__ float sv[DDIM];
  __shared__ float smax[16];
  __shared__ float smin[16];

  const int b = blockIdx.x;
  const int i = threadIdx.x;

  float kv = K[(size_t)b * DDIM + i];
  sk[i] = kv;
  sv[i] = V[(size_t)b * DDIM + i];

  // block-wide max/min of k row: wave shuffle reduce, then 16->1 in wave 0
  float mx = kv, mn = kv;
  #pragma unroll
  for (int off = 16; off > 0; off >>= 1) {
    mx = fmaxf(mx, __shfl_xor(mx, off, 32));
    mn = fminf(mn, __shfl_xor(mn, off, 32));
  }
  if ((i & 31) == 0) { smax[i >> 5] = mx; smin[i >> 5] = mn; }
  __syncthreads();
  if (i < 16) {
    mx = smax[i]; mn = smin[i];
    #pragma unroll
    for (int off = 8; off > 0; off >>= 1) {
      mx = fmaxf(mx, __shfl_xor(mx, off, 32));
      mn = fminf(mn, __shfl_xor(mn, off, 32));
    }
    if (i == 0) { smax[0] = mx; smin[0] = mn; }
  }
  __syncthreads();
  const float kmax = smax[0], kmin = smin[0];

  const float qi = Q[(size_t)b * DDIM + i];
  const float m  = fmaxf(qi * kmax, qi * kmin);  // exact max_j (qi*k_j)

  float num = 0.f, den = 0.f;
  #pragma unroll 4
  for (int j = 0; j < DDIM; ++j) {
    float e = __expf(fmaf(qi, sk[j], -m));
    num = fmaf(e, sv[j], num);
    den += e;
  }
  out[(size_t)b * DDIM + i] = num / den;
}

// ---------------------------------------------------------------------------
extern "C" void kernel_launch(void* const* d_in, const int* in_sizes, int n_in,
                              void* d_out, int out_size, void* d_ws, size_t ws_size,
                              hipStream_t stream) {
  (void)in_sizes; (void)n_in; (void)out_size; (void)ws_size;
  const float* x  = (const float*)d_in[0];
  const float* Wq = (const float*)d_in[1];
  const float* Wk = (const float*)d_in[2];
  const float* Wv = (const float*)d_in[3];

  // workspace layout (8.5 MB total, all offsets 256B aligned):
  //   [0, 1MB)        xb   : bf16 x
  //   [1MB, 2.5MB)    wts  : 3 x bf16 W^T
  //   [2.5MB, 8.5MB)  qkv  : 3 x f32 projections
  char* ws = (char*)d_ws;
  unsigned short* xb  = (unsigned short*)(ws);
  unsigned short* wts = (unsigned short*)(ws + (size_t)NROWS * DDIM * 2);
  float*          qkv = (float*)(ws + (size_t)NROWS * DDIM * 2
                                    + (size_t)3 * DDIM * DDIM * 2);

  convert_kernel<<<(NROWS * DDIM + 255) / 256, 256, 0, stream>>>(
      x, Wq, Wk, Wv, xb, wts);

  proj_gemm_wmma<<<dim3(NROWS / 16, DDIM / 16, 3), 32, 0, stream>>>(
      xb, wts, qkv);

  attn_kernel<<<NROWS, DDIM, 0, stream>>>(qkv, (float*)d_out);
}
